// lstm_cell_level_10660108829348
// MI455X (gfx1250) — matrix operand order repaired and checked by reading, not compiled, not run on hardware
//
#include <hip/hip_runtime.h>

#define HIDDEN 512
#define GATES 2048       // 4*HIDDEN
#define IN_FEAT 96
#define KTOT 608         // HIDDEN + IN_FEAT
#define B_SIZE 8192
#define T_STEPS 32
#define NUM_CLASSES 10
#define NCHUNK (KTOT / 32)      // 19 K-chunks of 32
#define LDS_ROW 40              // 32 K-halves padded to 40 (80B row stride, 16B aligned)
#define LDS_BUF (128 * LDS_ROW) // one slice: 128 rows x 40 halves = 10240 B

typedef __attribute__((ext_vector_type(16))) __bf16        v16bf;
typedef __attribute__((ext_vector_type(8)))  float         v8f;
typedef __attribute__((ext_vector_type(4)))  unsigned int  v4u;

union FragBF { v4u q[2]; v16bf v; };

__device__ __forceinline__ unsigned short f32_to_bf16(float f) {
    unsigned int u = __float_as_uint(f);
    u += 0x7FFFu + ((u >> 16) & 1u);           // round-to-nearest-even
    return (unsigned short)(u >> 16);
}
__device__ __forceinline__ float bf16_to_f32(unsigned short h) {
    return __uint_as_float(((unsigned int)h) << 16);
}
__device__ __forceinline__ float sigmoidf_(float x) {
    return 1.0f / (1.0f + __expf(-x));
}

// ---------------- prep: pack weights (bf16), bias sum, init h (bf16) and c ----------------
__global__ void prep_kernel(const float* __restrict__ W_ih, const float* __restrict__ W_hh,
                            const float* __restrict__ b_ih, const float* __restrict__ b_hh,
                            const float* __restrict__ h0,   const float* __restrict__ c0,
                            unsigned short* __restrict__ Wg, float* __restrict__ bias,
                            unsigned short* __restrict__ hA, float* __restrict__ cS) {
    int idx    = blockIdx.x * blockDim.x + threadIdx.x;
    int stride = gridDim.x * blockDim.x;
    const int NW = GATES * KTOT;
    for (int i = idx; i < NW; i += stride) {
        int g = i / KTOT, k = i % KTOT;
        float v = (k < HIDDEN) ? W_hh[(size_t)g * HIDDEN + k]
                               : W_ih[(size_t)g * IN_FEAT + (k - HIDDEN)];
        Wg[i] = f32_to_bf16(v);
    }
    for (int i = idx; i < GATES; i += stride) bias[i] = b_ih[i] + b_hh[i];
    const int NH = B_SIZE * HIDDEN;
    for (int i = idx; i < NH; i += stride) { hA[i] = f32_to_bf16(h0[i]); cS[i] = c0[i]; }
}

// ---------------- prep: gather image rows into xT[T][B][96] bf16 ----------------
__global__ void prep_x_kernel(const float* __restrict__ inputs, unsigned short* __restrict__ xT) {
    long long idx    = (long long)blockIdx.x * blockDim.x + threadIdx.x;
    long long stride = (long long)gridDim.x * blockDim.x;
    const long long N = (long long)T_STEPS * B_SIZE * IN_FEAT;
    for (long long i = idx; i < N; i += stride) {
        int t = (int)(i / ((long long)B_SIZE * IN_FEAT));
        int r = (int)(i % ((long long)B_SIZE * IN_FEAT));
        int b = r / IN_FEAT, f = r % IN_FEAT;
        int w = f / 3, c = f % 3;
        // inputs[b][c][h=t][w], strides b:3072 c:1024 h:32 w:1
        xT[i] = f32_to_bf16(inputs[(long long)b * 3072 + c * 1024 + t * 32 + w]);
    }
}

// ---------------- one LSTM timestep: gates GEMM (bf16 WMMA, LDS double-buffered B) ----------------
__launch_bounds__(256, 1)
__global__ void lstm_step_kernel(const unsigned short* __restrict__ hPrev,  // [B][512] bf16
                                 const unsigned short* __restrict__ xT_t,   // [B][96]  bf16
                                 const unsigned short* __restrict__ Wg,     // [2048][608] bf16
                                 const float* __restrict__ bias,            // [2048]
                                 float* __restrict__ cS,                    // [B][512]
                                 unsigned short* __restrict__ hNext) {      // [B][512] bf16
    __shared__ alignas(16) unsigned short ldsB[2 * LDS_BUF];    // 20 KB, double buffered

    const int tid     = threadIdx.x;
    const int lane    = tid & 31;
    const int wave    = tid >> 5;
    const int rowBase = blockIdx.x * 128 + wave * 16;   // batch rows this wave owns
    const int colBase = blockIdx.y * 32;                // hidden cols (per gate)
    const int laneN   = lane & 15;
    const int hiHalf  = lane >> 4;                      // 0 or 1 (lanes 16-31)
    const int hbA     = hiHalf * 8;                     // A-side K sub-offset (halves)
    const int rA      = rowBase + laneN;                // A row for this lane

    // cooperative B-slice staging: each thread owns (row, 32B segment)
    const int bRowL = tid & 127;                        // output unit within 128-row panel
    const int bSeg  = (tid >> 7) * 16;                  // 0 or 16 halves (32B granule)
    const unsigned short* wgRow =
        Wg + (size_t)(((bRowL >> 5) * HIDDEN) + colBase + (bRowL & 31)) * KTOT;

    const unsigned short* aH = hPrev + (size_t)rA * HIDDEN;
    const unsigned short* aX = xT_t  + (size_t)rA * IN_FEAT;

    // accumulators: [gate][colTile] of 16x16 f32, initialized with per-column bias
    v8f acc[4][2];
    #pragma unroll
    for (int g = 0; g < 4; ++g) {
        #pragma unroll
        for (int t2 = 0; t2 < 2; ++t2) {
            float bv = bias[g * HIDDEN + colBase + t2 * 16 + laneN];
            #pragma unroll
            for (int e = 0; e < 8; ++e) acc[g][t2][e] = bv;
        }
    }

    auto loadA = [&](int kc, FragBF& f) {
        if (kc < 16) {
            const unsigned short* p = aH + kc * 32;
            f.q[0] = *(const v4u*)(p + hbA);
            f.q[1] = *(const v4u*)(p + 16 + hbA);
        } else {
            const unsigned short* p = aX + (kc - 16) * 32;
            f.q[0] = *(const v4u*)(p + hbA);
            f.q[1] = *(const v4u*)(p + 16 + hbA);
        }
    };
    auto loadBslice = [&](int kc, v4u& s0, v4u& s1) {
        const unsigned short* p = wgRow + kc * 32 + bSeg;
        s0 = *(const v4u*)(p);
        s1 = *(const v4u*)(p + 8);
    };

    // prologue: chunk 0 into registers, stage B slice into LDS buf0
    FragBF aCur; loadA(0, aCur);
    {
        v4u s0, s1; loadBslice(0, s0, s1);
        *(v4u*)(&ldsB[bRowL * LDS_ROW + bSeg])     = s0;
        *(v4u*)(&ldsB[bRowL * LDS_ROW + bSeg + 8]) = s1;
    }
    __syncthreads();

    // steady state: compute chunk kc from buf[kc&1], prefetch+stage chunk kc+1 into buf[(kc+1)&1]
    for (int kc = 0; kc < NCHUNK - 1; ++kc) {
        const unsigned short* bufp = &ldsB[(kc & 1) * LDS_BUF];
        unsigned short*       bufn = &ldsB[((kc + 1) & 1) * LDS_BUF];

        FragBF aNext; v4u n0, n1;
        loadA(kc + 1, aNext);
        loadBslice(kc + 1, n0, n1);

        FragBF bf[4][2];
        #pragma unroll
        for (int g = 0; g < 4; ++g) {
            #pragma unroll
            for (int t2 = 0; t2 < 2; ++t2) {
                const unsigned short* lp = bufp + (g * 32 + t2 * 16 + laneN) * LDS_ROW + hiHalf * 8;
                bf[g][t2].q[0] = *(const v4u*)(lp);
                bf[g][t2].q[1] = *(const v4u*)(lp + 16);
            }
        }

        #pragma unroll
        for (int g = 0; g < 4; ++g) {
            #pragma unroll
            for (int t2 = 0; t2 < 2; ++t2) {
                acc[g][t2] = __builtin_amdgcn_wmma_f32_16x16x32_bf16(
                    false, aCur.v, false, bf[g][t2].v, (short)0, acc[g][t2], false, false);
            }
        }

        *(v4u*)(bufn + bRowL * LDS_ROW + bSeg)     = n0;
        *(v4u*)(bufn + bRowL * LDS_ROW + bSeg + 8) = n1;

        // pin the pipeline order: prefetch -> all DS reads -> WMMA burst -> DS writes
        __builtin_amdgcn_sched_group_barrier(0x020, 4, 0);   // 4 VMEM reads (prefetch)
        __builtin_amdgcn_sched_group_barrier(0x100, 16, 0);  // 16 DS reads (8 B frags)
        __builtin_amdgcn_sched_group_barrier(0x008, 8, 0);   // 8 WMMA
        __builtin_amdgcn_sched_group_barrier(0x200, 2, 0);   // 2 DS writes (stage next)

        __syncthreads();
        aCur = aNext;
    }

    // final chunk (kc = NCHUNK-1 = 18, slice lives in buf[0])
    {
        const unsigned short* bufp = &ldsB[((NCHUNK - 1) & 1) * LDS_BUF];
        FragBF bf[4][2];
        #pragma unroll
        for (int g = 0; g < 4; ++g) {
            #pragma unroll
            for (int t2 = 0; t2 < 2; ++t2) {
                const unsigned short* lp = bufp + (g * 32 + t2 * 16 + laneN) * LDS_ROW + hiHalf * 8;
                bf[g][t2].q[0] = *(const v4u*)(lp);
                bf[g][t2].q[1] = *(const v4u*)(lp + 16);
            }
        }
        #pragma unroll
        for (int g = 0; g < 4; ++g) {
            #pragma unroll
            for (int t2 = 0; t2 < 2; ++t2) {
                acc[g][t2] = __builtin_amdgcn_wmma_f32_16x16x32_bf16(
                    false, aCur.v, false, bf[g][t2].v, (short)0, acc[g][t2], false, false);
            }
        }
        __builtin_amdgcn_sched_group_barrier(0x100, 16, 0);
        __builtin_amdgcn_sched_group_barrier(0x008, 8, 0);
    }

    // ---- gating epilogue: C/D layout -> row = rowBase + e + 8*hiHalf, col = colBase + t2*16 + laneN
    #pragma unroll
    for (int t2 = 0; t2 < 2; ++t2) {
        int col = colBase + t2 * 16 + laneN;
        #pragma unroll
        for (int e = 0; e < 8; ++e) {
            int row = rowBase + e + hiHalf * 8;
            float gi = sigmoidf_(acc[0][t2][e]);
            float gf = sigmoidf_(acc[1][t2][e]);
            float gg = tanhf(acc[2][t2][e]);
            float go = sigmoidf_(acc[3][t2][e]);
            size_t cidx = (size_t)row * HIDDEN + col;
            float cOld = cS[cidx];
            float cNew = gf * cOld + gi * gg;
            cS[cidx] = cNew;
            hNext[cidx] = f32_to_bf16(go * tanhf(cNew));
        }
    }
}

// ---------------- final projection: out[B][10] = h·W_outT + b_out ----------------
__global__ void out_kernel(const unsigned short* __restrict__ hFin,
                           const float* __restrict__ W_out, const float* __restrict__ b_out,
                           float* __restrict__ out) {
    int idx = blockIdx.x * blockDim.x + threadIdx.x;
    if (idx >= B_SIZE * NUM_CLASSES) return;
    int b = idx / NUM_CLASSES, n = idx % NUM_CLASSES;
    const unsigned short* h = hFin + (size_t)b * HIDDEN;
    const float* w = W_out + (size_t)n * HIDDEN;
    float s = b_out[n];
    #pragma unroll 8
    for (int k = 0; k < HIDDEN; ++k) s += bf16_to_f32(h[k]) * w[k];
    out[idx] = s;
}

// ---------------- workspace layout (bytes) ----------------
static constexpr size_t WG_BYTES   = (size_t)GATES * KTOT * 2;              // 2,490,368
static constexpr size_t BIAS_BYTES = (size_t)GATES * 4;                     // 8,192
static constexpr size_t X_BYTES    = (size_t)T_STEPS * B_SIZE * IN_FEAT * 2;// 50,331,648
static constexpr size_t H_BYTES    = (size_t)B_SIZE * HIDDEN * 2;           // 8,388,608
static constexpr size_t C_BYTES    = (size_t)B_SIZE * HIDDEN * 4;           // 16,777,216

static constexpr size_t WG_OFF   = 0;
static constexpr size_t BIAS_OFF = WG_OFF + WG_BYTES;
static constexpr size_t X_OFF    = BIAS_OFF + BIAS_BYTES;
static constexpr size_t HA_OFF   = X_OFF + X_BYTES;
static constexpr size_t HB_OFF   = HA_OFF + H_BYTES;
static constexpr size_t C_OFF    = HB_OFF + H_BYTES;   // total ~86.4 MB

extern "C" void kernel_launch(void* const* d_in, const int* in_sizes, int n_in,
                              void* d_out, int out_size, void* d_ws, size_t ws_size,
                              hipStream_t stream) {
    const float* inputs = (const float*)d_in[0];
    const float* h0     = (const float*)d_in[1];
    const float* c0     = (const float*)d_in[2];
    const float* W_ih   = (const float*)d_in[3];
    const float* W_hh   = (const float*)d_in[4];
    const float* b_ih   = (const float*)d_in[5];
    const float* b_hh   = (const float*)d_in[6];
    const float* W_out  = (const float*)d_in[7];
    const float* b_out  = (const float*)d_in[8];
    float* out = (float*)d_out;

    char* ws = (char*)d_ws;
    unsigned short* Wg   = (unsigned short*)(ws + WG_OFF);
    float*          bias = (float*)(ws + BIAS_OFF);
    unsigned short* xT   = (unsigned short*)(ws + X_OFF);
    unsigned short* hA   = (unsigned short*)(ws + HA_OFF);
    unsigned short* hB   = (unsigned short*)(ws + HB_OFF);
    float*          cS   = (float*)(ws + C_OFF);

    prep_kernel<<<4096, 256, 0, stream>>>(W_ih, W_hh, b_ih, b_hh, h0, c0, Wg, bias, hA, cS);
    prep_x_kernel<<<8192, 256, 0, stream>>>(inputs, xT);

    dim3 grid(B_SIZE / 128, HIDDEN / 32);   // 64 x 16 blocks, 256 threads each
    for (int t = 0; t < T_STEPS; ++t) {
        const unsigned short* hPrev = (t & 1) ? hB : hA;
        unsigned short*       hNext = (t & 1) ? hA : hB;
        lstm_step_kernel<<<grid, 256, 0, stream>>>(
            hPrev, xT + (size_t)t * B_SIZE * IN_FEAT, Wg, bias, cS, hNext);
    }
    // 32 steps: last write (t=31, odd) lands in hA
    out_kernel<<<(B_SIZE * NUM_CLASSES + 255) / 256, 256, 0, stream>>>(hA, W_out, b_out, out);
}
